// WindowDecoder_14345190769162
// MI455X (gfx1250) — compile-verified, matrix operand order checked by reading, not run on hardware
//
#include <hip/hip_runtime.h>
#include <hip/hip_bf16.h>

// ---------------------------------------------------------------------------
// WindowDecoder on MI455X (gfx1250, wave32, WMMA).
//
// Pipeline:   h0 = tanh(z @ w_emb^T + b_emb)                [8192, 256]
//   s=0..19:  gh = h @ w_hh^T + b_hh ; GRU gate update      (row-independent)
//             y  = relu(h @ w_mlp^T + b_mlp)  (x2, same W)
//             out[n*20+s, :] = y @ w_mu^T + b_mu            [.., 64]
//
// One wave32 owns a 16-row tile end-to-end (persistent); h/y tiles in LDS
// (fp32); all GEMMs via v_wmma_f32_16x16x32_f16 (f32 accumulate); h0 staged
// into LDS with global_load_async_to_lds_b128; gates use hardware V_TANH_F32.
// ---------------------------------------------------------------------------

typedef __attribute__((ext_vector_type(16))) _Float16 v16h;
typedef __attribute__((ext_vector_type(8)))  float    v8f;
typedef __attribute__((ext_vector_type(4)))  int      v4i;

#define HID   256      // hidden size H
#define EMB_K 128      // L + G
#define OUTD  64       // output size O
#define STEPS 20       // output_len
#define NROWS 8192     // B * T
#define WAVES_PER_BLK 4
#define ROWS_PER_WAVE 16
#define NBLOCKS (NROWS / (WAVES_PER_BLK * ROWS_PER_WAVE))   // 128

#define AS1 __attribute__((address_space(1)))
#define AS3 __attribute__((address_space(3)))

union BFrag { v16h v; uint4 q[2]; };

static __device__ __forceinline__ v8f wmma_f16(v16h a, v16h b, v8f c) {
  // D = A(16x32 f16) * B(32x16 f16) + C(16x16 f32)
  return __builtin_amdgcn_wmma_f32_16x16x32_f16(false, a, false, b, (short)0, c,
                                                false, false);
}

// B fragment (K=32 x N=16) from a row-major f16 weight matrix W[N_total][ldk]:
// element (k, n): n = lane%16, k = kb + 16*(e/8) + 8*(lane/16) + e%8.
// Per lane: two contiguous 16B chunks -> global_load_b128 x2.
static __device__ __forceinline__ v16h load_b_f16(const _Float16* __restrict__ W,
                                                  int ldk, int nbase, int kb,
                                                  int lane) {
  BFrag f;
  const _Float16* p =
      W + (size_t)(nbase + (lane & 15)) * ldk + kb + ((lane >> 4) << 3);
  f.q[0] = *(const uint4*)(p);
  f.q[1] = *(const uint4*)(p + 16);
  return f.v;
}

// A fragment (M=16 x K=32) from an LDS fp32 tile X[16][HID] (convert to f16).
static __device__ __forceinline__ v16h load_a_lds(const float* __restrict__ X,
                                                  int kb, int lane) {
  const float* p = X + (lane & 15) * HID + kb + ((lane >> 4) << 3);
  float4 c0 = *(const float4*)(p);
  float4 c1 = *(const float4*)(p + 4);
  float4 c2 = *(const float4*)(p + 16);
  float4 c3 = *(const float4*)(p + 20);
  float tmp[16] = {c0.x, c0.y, c0.z, c0.w, c1.x, c1.y, c1.z, c1.w,
                   c2.x, c2.y, c2.z, c2.w, c3.x, c3.y, c3.z, c3.w};
  v16h a;
#pragma unroll
  for (int j = 0; j < 16; ++j) a[j] = (_Float16)tmp[j];
  return a;
}

// ---- activations: prefer hardware V_TANH_F32 (gfx1250 tanh-insts) ----
static __device__ __forceinline__ float tanh_hw(float x) {
#if __has_builtin(__builtin_amdgcn_tanhf)
  return __builtin_amdgcn_tanhf(x);
#elif __has_builtin(__builtin_amdgcn_rcpf)
  // tanh(x) = 1 - 2/(exp(2x)+1), fast rcp (no IEEE div expansion)
  return 1.0f - 2.0f * __builtin_amdgcn_rcpf(__expf(2.0f * x) + 1.0f);
#else
  return tanhf(x);
#endif
}
static __device__ __forceinline__ float sigmoid_hw(float x) {
  // sigmoid(x) = 0.5 * tanh(x/2) + 0.5  -> single transcendental
  return __builtin_fmaf(0.5f, tanh_hw(0.5f * x), 0.5f);
}

// ---- async global->LDS staging (ASYNCcnt path) ----
static __device__ __forceinline__ void wait_async0() {
#if __has_builtin(__builtin_amdgcn_s_wait_asynccnt)
  __builtin_amdgcn_s_wait_asynccnt(0);
#else
  asm volatile("s_wait_asynccnt 0" ::: "memory");
#endif
}

static __device__ __forceinline__ void stage_tile_to_lds(
    const float* __restrict__ src, float* __restrict__ lds, int lane) {
  // 16 rows x 256 f32 = 16 KB, 16 B per lane per iteration.
#if __has_builtin(__builtin_amdgcn_global_load_async_to_lds_b128)
#pragma unroll 1
  for (int i = lane; i < ROWS_PER_WAVE * HID / 4; i += 32) {
    __builtin_amdgcn_global_load_async_to_lds_b128(
        (AS1 v4i*)((const float4*)src + i), (AS3 v4i*)((float4*)lds + i),
        0, 0);
  }
  wait_async0();
#else
  const float4* s4 = (const float4*)src;
  float4* d4 = (float4*)lds;
#pragma unroll 1
  for (int i = lane; i < ROWS_PER_WAVE * HID / 4; i += 32) d4[i] = s4[i];
#endif
}

// -------------------------------- kernels ----------------------------------

__global__ void cvt_f32_f16_kernel(const float* __restrict__ src,
                                   _Float16* __restrict__ dst, int n) {
  int i = blockIdx.x * blockDim.x + threadIdx.x;
  if (i < n) dst[i] = (_Float16)src[i];
}

// h0 = tanh(z @ w_emb^T + b_emb); z = [z_t | z_g], K = 128, N = 256.
__global__ __launch_bounds__(128) void embed_kernel(
    const float* __restrict__ zt, const float* __restrict__ zg,
    const _Float16* __restrict__ wemb, const float* __restrict__ bemb,
    float* __restrict__ h0) {
  const int lane = threadIdx.x & 31;
  const int wave = threadIdx.x >> 5;
  const int rowbase = (blockIdx.x * WAVES_PER_BLK + wave) * ROWS_PER_WAVE;
  const int m = lane & 15;
  const int half8 = (lane >> 4) << 3;   // 0 or 8 (both K-split and M-split)
  const int n = rowbase + m;

  // A fragments straight from global z (K tiles 0,32 from z_t; 64,96 from z_g)
  v16h afr[4];
#pragma unroll
  for (int i = 0; i < 4; ++i) {
    const float* src = (i < 2)
        ? (zt + (size_t)n * 64 + i * 32 + half8)
        : (zg + (size_t)(n >> 6) * 64 + (i - 2) * 32 + half8);
    float4 c0 = *(const float4*)(src);
    float4 c1 = *(const float4*)(src + 4);
    float4 c2 = *(const float4*)(src + 16);
    float4 c3 = *(const float4*)(src + 20);
    float tmp[16] = {c0.x, c0.y, c0.z, c0.w, c1.x, c1.y, c1.z, c1.w,
                     c2.x, c2.y, c2.z, c2.w, c3.x, c3.y, c3.z, c3.w};
    v16h a;
#pragma unroll
    for (int j = 0; j < 16; ++j) a[j] = (_Float16)tmp[j];
    afr[i] = a;
  }

#pragma unroll 1
  for (int t = 0; t < HID / 16; ++t) {
    v8f acc = {0.f, 0.f, 0.f, 0.f, 0.f, 0.f, 0.f, 0.f};
#pragma unroll
    for (int i = 0; i < 4; ++i)
      acc = wmma_f16(afr[i], load_b_f16(wemb, EMB_K, t * 16, i * 32, lane), acc);
    const int col = t * 16 + m;
    const float bv = bemb[col];
#pragma unroll
    for (int v = 0; v < 8; ++v) {
      const int mr = v + half8;
      h0[(size_t)(rowbase + mr) * HID + col] = tanh_hw(acc[v] + bv);
    }
  }
}

// Persistent per-tile kernel: 20 GRU steps + MLP x2 + mu, writing d_out.
__global__ __launch_bounds__(128) void gru_mlp_kernel(
    const float* __restrict__ h0, const _Float16* __restrict__ whh,
    const _Float16* __restrict__ wmlp, const _Float16* __restrict__ wmu,
    const float* __restrict__ bih, const float* __restrict__ bhh,
    const float* __restrict__ bmlp, const float* __restrict__ bmu,
    float* __restrict__ out) {
  extern __shared__ float smem[];   // [2][WAVES][16][HID] fp32 = 128 KB
  const int lane = threadIdx.x & 31;
  const int wave = threadIdx.x >> 5;
  const int rowbase = (blockIdx.x * WAVES_PER_BLK + wave) * ROWS_PER_WAVE;
  float* hb = smem + (size_t)wave * ROWS_PER_WAVE * HID;
  float* yb = smem + (size_t)(WAVES_PER_BLK + wave) * ROWS_PER_WAVE * HID;
  const int m = lane & 15;
  const int half8 = (lane >> 4) << 3;

  // Warm GL2 with the GRU weight matrix (streamed 20x per tile).
#pragma unroll 1
  for (int i = threadIdx.x * 64; i < 3 * HID * HID; i += blockDim.x * 64)
    __builtin_prefetch(whh + i, 0, 1);

  // Stage this wave's 16 h-rows into LDS (async DMA path, ASYNCcnt).
  stage_tile_to_lds(h0 + (size_t)rowbase * HID, hb, lane);
  __syncthreads();

  v16h afr[8];

#pragma unroll 1
  for (int s = 0; s < STEPS; ++s) {
    // ---- GRU step: gh = h @ w_hh^T, gate update in-place on hb ----
#pragma unroll
    for (int i = 0; i < 8; ++i) afr[i] = load_a_lds(hb, i * 32, lane);
    __syncthreads();
#pragma unroll 1
    for (int t = 0; t < HID / 16; ++t) {
      v8f ar = {0.f, 0.f, 0.f, 0.f, 0.f, 0.f, 0.f, 0.f};
      v8f az = ar, an = ar;
#pragma unroll
      for (int i = 0; i < 8; ++i) {
        const int kb = i * 32;
        ar = wmma_f16(afr[i], load_b_f16(whh, HID, t * 16,           kb, lane), ar);
        az = wmma_f16(afr[i], load_b_f16(whh, HID, HID + t * 16,     kb, lane), az);
        an = wmma_f16(afr[i], load_b_f16(whh, HID, 2 * HID + t * 16, kb, lane), an);
      }
      const int j = t * 16 + m;
      const float gir = bih[j], giz = bih[HID + j], gin = bih[2 * HID + j];
      const float bhr = bhh[j], bhz = bhh[HID + j], bhn = bhh[2 * HID + j];
#pragma unroll
      for (int v = 0; v < 8; ++v) {
        const int mr = v + half8;
        const float r  = sigmoid_hw(gir + ar[v] + bhr);
        const float u  = sigmoid_hw(giz + az[v] + bhz);
        const float nn = tanh_hw(gin + r * (an[v] + bhn));
        const float hold = hb[mr * HID + j];
        hb[mr * HID + j] = (1.0f - u) * nn + u * hold;
      }
    }
    __syncthreads();

    // ---- MLP layer 1: yb = relu(hb @ w_mlp^T + b_mlp) ----
#pragma unroll
    for (int i = 0; i < 8; ++i) afr[i] = load_a_lds(hb, i * 32, lane);
    __syncthreads();
#pragma unroll 1
    for (int t = 0; t < HID / 16; ++t) {
      v8f acc = {0.f, 0.f, 0.f, 0.f, 0.f, 0.f, 0.f, 0.f};
#pragma unroll
      for (int i = 0; i < 8; ++i)
        acc = wmma_f16(afr[i], load_b_f16(wmlp, HID, t * 16, i * 32, lane), acc);
      const int j = t * 16 + m;
      const float bv = bmlp[j];
#pragma unroll
      for (int v = 0; v < 8; ++v)
        yb[(v + half8) * HID + j] = fmaxf(acc[v] + bv, 0.0f);
    }
    __syncthreads();

    // ---- MLP layer 2 (same weights), in-place on yb ----
#pragma unroll
    for (int i = 0; i < 8; ++i) afr[i] = load_a_lds(yb, i * 32, lane);
    __syncthreads();
#pragma unroll 1
    for (int t = 0; t < HID / 16; ++t) {
      v8f acc = {0.f, 0.f, 0.f, 0.f, 0.f, 0.f, 0.f, 0.f};
#pragma unroll
      for (int i = 0; i < 8; ++i)
        acc = wmma_f16(afr[i], load_b_f16(wmlp, HID, t * 16, i * 32, lane), acc);
      const int j = t * 16 + m;
      const float bv = bmlp[j];
#pragma unroll
      for (int v = 0; v < 8; ++v)
        yb[(v + half8) * HID + j] = fmaxf(acc[v] + bv, 0.0f);
    }
    __syncthreads();

    // ---- mu head: out[(n*20+s), :] = yb @ w_mu^T + b_mu ----
#pragma unroll
    for (int i = 0; i < 8; ++i) afr[i] = load_a_lds(yb, i * 32, lane);
    __syncthreads();
#pragma unroll 1
    for (int t = 0; t < OUTD / 16; ++t) {
      v8f acc = {0.f, 0.f, 0.f, 0.f, 0.f, 0.f, 0.f, 0.f};
#pragma unroll
      for (int i = 0; i < 8; ++i)
        acc = wmma_f16(afr[i], load_b_f16(wmu, HID, t * 16, i * 32, lane), acc);
      const int j = t * 16 + m;
      const float bv = bmu[j];
#pragma unroll
      for (int v = 0; v < 8; ++v) {
        const int mr = v + half8;
        out[((size_t)(rowbase + mr) * STEPS + s) * OUTD + j] = acc[v] + bv;
      }
    }
    __syncthreads();   // hb rewritten next step; keep phases clean
  }
}

// ------------------------------ host launch --------------------------------

extern "C" void kernel_launch(void* const* d_in, const int* in_sizes, int n_in,
                              void* d_out, int out_size, void* d_ws, size_t ws_size,
                              hipStream_t stream) {
  const float* z_t   = (const float*)d_in[0];
  const float* z_g   = (const float*)d_in[1];
  const float* w_emb = (const float*)d_in[2];
  const float* b_emb = (const float*)d_in[3];
  // d_in[4] = w_ih: unused (GRU input is all-zeros; only b_ih survives)
  const float* b_ih  = (const float*)d_in[5];
  const float* w_hh  = (const float*)d_in[6];
  const float* b_hh  = (const float*)d_in[7];
  const float* w_mlp = (const float*)d_in[8];
  const float* b_mlp = (const float*)d_in[9];
  const float* w_mu  = (const float*)d_in[10];
  const float* b_mu  = (const float*)d_in[11];
  float* out = (float*)d_out;

  // Workspace layout: fp32 h0 [8192x256], then f16 weight copies.
  float* h0 = (float*)d_ws;
  _Float16* wemb16 = (_Float16*)((char*)d_ws + (size_t)NROWS * HID * sizeof(float));
  _Float16* whh16  = wemb16 + HID * EMB_K;
  _Float16* wmlp16 = whh16 + 3 * HID * HID;
  _Float16* wmu16  = wmlp16 + HID * HID;

  auto cvt = [&](const float* s, _Float16* d, int n) {
    cvt_f32_f16_kernel<<<(n + 255) / 256, 256, 0, stream>>>(s, d, n);
  };
  cvt(w_emb, wemb16, HID * EMB_K);
  cvt(w_hh,  whh16,  3 * HID * HID);
  cvt(w_mlp, wmlp16, HID * HID);
  cvt(w_mu,  wmu16,  OUTD * HID);

  embed_kernel<<<NBLOCKS, 128, 0, stream>>>(z_t, z_g, wemb16, b_emb, h0);

  const size_t smem_bytes = 2u * WAVES_PER_BLK * ROWS_PER_WAVE * HID * sizeof(float);
  gru_mlp_kernel<<<NBLOCKS, 128, smem_bytes, stream>>>(
      h0, whh16, wmlp16, wmu16, b_ih, b_hh, b_mlp, b_mu, out);
}
